// DeepGraphTransformer_42966852829453
// MI455X (gfx1250) — matrix operand order repaired and checked by reading, not compiled, not run on hardware
//
#include <hip/hip_runtime.h>
#include <math.h>

#define NN 10000
#define NE 160000
#define NG 64
#define HID 128
#define LAYERS 6

typedef __attribute__((ext_vector_type(16))) __bf16        v16bf;
typedef __attribute__((ext_vector_type(8)))  unsigned int  v8u;
typedef __attribute__((ext_vector_type(4)))  unsigned int  v4u;
typedef __attribute__((ext_vector_type(8)))  float         v8f;

// ---------- device helpers ----------
__device__ __forceinline__ unsigned short f32_to_bf16(float f) {
    unsigned int u = __float_as_uint(f);
    u += 0x7FFFu + ((u >> 16) & 1u);          // round-to-nearest-even
    return (unsigned short)(u >> 16);
}
__device__ __forceinline__ unsigned int pack_bf16x2(float lo, float hi) {
    return (unsigned int)f32_to_bf16(lo) | ((unsigned int)f32_to_bf16(hi) << 16);
}
__device__ __forceinline__ float bf16_to_f32(unsigned short h) {
    return __uint_as_float(((unsigned int)h) << 16);
}
__device__ __forceinline__ float wave_sum(float v) {
#pragma unroll
    for (int off = 16; off; off >>= 1) v += __shfl_xor(v, off, 32);
    return v;
}
__device__ __forceinline__ float geluf(float x) {
    return 0.5f * x * (1.0f + erff(x * 0.7071067811865476f));
}
// float atomic max via monotonic integer-bits mapping
__device__ __forceinline__ void atomic_max_f32(float* a, float v) {
    if (v >= 0.0f) atomicMax((int*)a, __float_as_int(v));
    else           atomicMin((unsigned int*)a, (unsigned int)__float_as_int(v));
}

// ---------- generic fill / casts ----------
__global__ __launch_bounds__(256) void fill_f32(float* p, float v, int n) {
    int i = blockIdx.x * 256 + threadIdx.x;
    if (i < n) p[i] = v;
}
__global__ __launch_bounds__(256) void cast_bf16(const float* __restrict__ s,
                                                 unsigned short* __restrict__ d, int n) {
    int i = blockIdx.x * 256 + threadIdx.x;
    if (i < n) d[i] = f32_to_bf16(s[i]);
}
// W[K,N] fp32 row-major -> WT[N,K] bf16 row-major (column-major B); K = 1<<kshift
__global__ __launch_bounds__(256) void cast_tr_bf16(const float* __restrict__ W,
                                                    unsigned short* __restrict__ WT,
                                                    int kshift, int N) {
    int i = blockIdx.x * 256 + threadIdx.x;
    if (i >= (N << kshift)) return;
    int n = i >> kshift, k = i & ((1 << kshift) - 1);
    WT[i] = f32_to_bf16(W[(size_t)k * N + n]);
}

// ---------- WMMA GEMM (LDS-free, all-bf16 operands) ----------
// C[M,N] = act(A[M,K] * B[K,N] + bias); A bf16 row-major, BT = B^T bf16 [N,K].
// Block: 256 thr = 8 waves; tile 32(M) x 128(N); wave = 16x32 (two 16x16 C tiles).
// Fragments load as contiguous 16B/32B global loads in a contiguous-k order
// (k = 16*hi + s, same in A and B); K templated so the loop fully unrolls into a
// straight run of v_wmma. Out-of-range A rows are CLAMPED (not zeroed): D-row m
// depends only on A-row m, and OOB rows are masked at the store, so the inner
// loop stays branch-free with no exec masking.
// OUTBF: 0 -> f32 out, 1 -> bf16 out.  ACT: 0 none, 1 gelu(erf), 2 relu.
template <int K, int OUTBF, int ACT>
__global__ __launch_bounds__(256)
void gemm_wmma(const unsigned short* __restrict__ A, const unsigned short* __restrict__ BT,
               const float* __restrict__ bias, void* __restrict__ C,
               int M, int N) {
    const int lane = threadIdx.x & 31;
    const int wave = threadIdx.x >> 5;
    const int l    = lane & 15;
    const int hi   = lane >> 4;
    const int wr   = wave >> 2;              // 0..1
    const int wc   = wave & 3;               // 0..3
    const int tileM = blockIdx.y * 32 + wr * 16;
    const int tileN = blockIdx.x * 128 + wc * 32;

    const int gra = min(tileM + l, M - 1);   // A row held by this lane (clamped)
    const unsigned short* aptr = A  + (size_t)gra * K;
    const unsigned short* bp0  = BT + (size_t)(tileN + l) * K;
    const unsigned short* bp1  = BT + (size_t)(tileN + 16 + l) * K;
    __builtin_prefetch(bp0, 0, 1);           // weights: prime GL2 (global_prefetch_b8)
    __builtin_prefetch(bp1, 0, 1);

    v8f acc0 = {0.f, 0.f, 0.f, 0.f, 0.f, 0.f, 0.f, 0.f};
    v8f acc1 = acc0;

#pragma unroll
    for (int kk = 0; kk < K; kk += 32) {
        // A fragment 16x32: lane holds K hi*16..+15 (contiguous) of its row
        v4u alo = *(const v4u*)(aptr + kk + hi * 16);
        v4u ahi = *(const v4u*)(aptr + kk + hi * 16 + 8);
        v8u au;
        au[0] = alo[0]; au[1] = alo[1]; au[2] = alo[2]; au[3] = alo[3];
        au[4] = ahi[0]; au[5] = ahi[1]; au[6] = ahi[2]; au[7] = ahi[3];
        // B fragments 32x16: lane = column, elements = 16 contiguous K values
        v8u bu0 = *(const v8u*)(bp0 + kk + hi * 16);
        v8u bu1 = *(const v8u*)(bp1 + kk + hi * 16);

        v16bf a = __builtin_bit_cast(v16bf, au);
        acc0 = __builtin_amdgcn_wmma_f32_16x16x32_bf16(
            false, a, false, __builtin_bit_cast(v16bf, bu0), (short)0, acc0, false, false);
        acc1 = __builtin_amdgcn_wmma_f32_16x16x32_bf16(
            false, a, false, __builtin_bit_cast(v16bf, bu1), (short)0, acc1, false, false);
    }

    // C/D layout: lane<16 -> M = r0, N = l ; lane>=16 -> M = 8+r0, N = l
#pragma unroll
    for (int r0 = 0; r0 < 8; ++r0) {
        int row = tileM + hi * 8 + r0;
        if (row >= M) continue;
#pragma unroll
        for (int t = 0; t < 2; ++t) {
            int col = tileN + t * 16 + l;
            float v = (t ? acc1[r0] : acc0[r0]) + bias[col];
            if (ACT == 1) v = geluf(v);
            if (ACT == 2) v = fmaxf(v, 0.0f);
            if (OUTBF) ((unsigned short*)C)[(size_t)row * N + col] = f32_to_bf16(v);
            else       ((float*)C)[(size_t)row * N + col] = v;
        }
    }
}

// ---------- input projection: h = x[N,4] @ Win[4,128] + bin (f32 + bf16 mirror) ----------
__global__ __launch_bounds__(256) void input_proj(const float* __restrict__ x,
                                                  const float* __restrict__ Win,
                                                  const float* __restrict__ bin,
                                                  float* __restrict__ h,
                                                  unsigned short* __restrict__ hb, int n) {
    int i = blockIdx.x * 256 + threadIdx.x;
    if (i >= n * HID) return;
    int r = i >> 7, c = i & 127;
    float acc = bin[c];
#pragma unroll
    for (int j = 0; j < 4; ++j) acc += x[r * 4 + j] * Win[j * HID + c];
    h[i] = acc;
    hb[i] = f32_to_bf16(acc);
}

// ---------- edge logits: alpha[e,h] = q[dst]·k[src] / sqrt(128); atomic segment max ----------
__global__ __launch_bounds__(256) void edge_logits(const unsigned short* __restrict__ q,
                                                   const unsigned short* __restrict__ k,
                                                   const int* __restrict__ src,
                                                   const int* __restrict__ dst,
                                                   float* __restrict__ alpha,
                                                   float* __restrict__ m, int E) {
    int e = blockIdx.x * 8 + (threadIdx.x >> 5);
    int lane = threadIdx.x & 31;
    if (e >= E) return;
    int sN = src[e], dN = dst[e];
    const unsigned short* qr = q + (size_t)dN * 1024;
    const unsigned short* kr = k + (size_t)sN * 1024;
#pragma unroll
    for (int h = 0; h < 8; ++h) {
        int base = h * 128 + lane * 4;
        float p = 0.f;
#pragma unroll
        for (int j = 0; j < 4; ++j) p += bf16_to_f32(qr[base + j]) * bf16_to_f32(kr[base + j]);
        p = wave_sum(p);
        if (lane == 0) {
            p *= 0.08838834764831845f;   // 1/sqrt(128)
            alpha[(size_t)e * 8 + h] = p;
            atomic_max_f32(&m[dN * 8 + h], p);
        }
    }
}

// ---------- softmax numerator + segment sum ----------
__global__ __launch_bounds__(256) void edge_exp(float* __restrict__ alpha,
                                                const float* __restrict__ m,
                                                float* __restrict__ s,
                                                const int* __restrict__ dst, int E) {
    int i = blockIdx.x * 256 + threadIdx.x;
    if (i >= E * 8) return;
    int e = i >> 3, h = i & 7;
    int d = dst[e];
    float v = __expf(alpha[i] - m[d * 8 + h]);
    alpha[i] = v;
    atomicAdd(&s[d * 8 + h], v);
}

// ---------- weighted scatter-add of V, head-mean folded (x0.125) ----------
__global__ __launch_bounds__(256) void edge_scatter(const float* __restrict__ alpha,
                                                    const float* __restrict__ s,
                                                    const unsigned short* __restrict__ v,
                                                    const int* __restrict__ src,
                                                    const int* __restrict__ dst,
                                                    float* __restrict__ outp, int E) {
    int e = blockIdx.x * 8 + (threadIdx.x >> 5);
    int lane = threadIdx.x & 31;
    if (e >= E) return;
    int sN = src[e], dN = dst[e];
    float a[8];
#pragma unroll
    for (int h = 0; h < 8; ++h) a[h] = alpha[(size_t)e * 8 + h] / s[dN * 8 + h];
    const unsigned short* vr = v + (size_t)sN * 1024;
    int c0 = lane * 4;
    float acc[4] = {0.f, 0.f, 0.f, 0.f};
#pragma unroll
    for (int h = 0; h < 8; ++h) {
        int b = h * 128 + c0;
#pragma unroll
        for (int j = 0; j < 4; ++j) acc[j] += a[h] * bf16_to_f32(vr[b + j]);
    }
    float* o = outp + (size_t)dN * HID + c0;
#pragma unroll
    for (int j = 0; j < 4; ++j) atomicAdd(o + j, acc[j] * 0.125f);
}

// ---------- beta gate + gelu + residual + layernorm (one wave / node) ----------
__global__ __launch_bounds__(256) void beta_ln(const float* __restrict__ outp,
                                               const float* __restrict__ xr,
                                               const float* __restrict__ Wb,
                                               const float* __restrict__ g1,
                                               const float* __restrict__ n1,
                                               float* __restrict__ h,
                                               unsigned short* __restrict__ hb, int n) {
    int nn = blockIdx.x * 8 + (threadIdx.x >> 5);
    int lane = threadIdx.x & 31;
    if (nn >= n) return;
    int c0 = lane * 4;
    float o[4], r[4], hv[4];
    float dot = 0.f;
#pragma unroll
    for (int j = 0; j < 4; ++j) {
        int c = c0 + j;
        o[j] = outp[(size_t)nn * HID + c];
        r[j] = xr[(size_t)nn * HID + c];
        hv[j] = h[(size_t)nn * HID + c];
        dot += o[j] * Wb[c] + r[j] * Wb[128 + c] + (o[j] - r[j]) * Wb[256 + c];
    }
    dot = wave_sum(dot);
    float beta = 1.0f / (1.0f + __expf(-dot));
    float gv[4], sum = 0.f, sq = 0.f;
#pragma unroll
    for (int j = 0; j < 4; ++j) {
        float cv = beta * r[j] + (1.0f - beta) * o[j];
        float g = geluf(cv) + hv[j];
        gv[j] = g; sum += g; sq += g * g;
    }
    sum = wave_sum(sum); sq = wave_sum(sq);
    float mu = sum * (1.0f / 128.0f);
    float inv = rsqrtf(sq * (1.0f / 128.0f) - mu * mu + 1e-5f);
    float outv[4];
#pragma unroll
    for (int j = 0; j < 4; ++j) {
        int c = c0 + j;
        outv[j] = (gv[j] - mu) * inv * g1[c] + n1[c];
        h[(size_t)nn * HID + c] = outv[j];
    }
    unsigned int* hb32 = (unsigned int*)hb;
    size_t p = ((size_t)nn * HID + c0) >> 1;
    hb32[p]     = pack_bf16x2(outv[0], outv[1]);
    hb32[p + 1] = pack_bf16x2(outv[2], outv[3]);
}

// ---------- residual + layernorm (post-FFN) ----------
__global__ __launch_bounds__(256) void residual_ln(const float* __restrict__ f,
                                                   const float* __restrict__ g2,
                                                   const float* __restrict__ n2,
                                                   float* __restrict__ h,
                                                   unsigned short* __restrict__ hb, int n) {
    int nn = blockIdx.x * 8 + (threadIdx.x >> 5);
    int lane = threadIdx.x & 31;
    if (nn >= n) return;
    int c0 = lane * 4;
    float gv[4], sum = 0.f, sq = 0.f;
#pragma unroll
    for (int j = 0; j < 4; ++j) {
        float g = f[(size_t)nn * HID + c0 + j] + h[(size_t)nn * HID + c0 + j];
        gv[j] = g; sum += g; sq += g * g;
    }
    sum = wave_sum(sum); sq = wave_sum(sq);
    float mu = sum * (1.0f / 128.0f);
    float inv = rsqrtf(sq * (1.0f / 128.0f) - mu * mu + 1e-5f);
    float outv[4];
#pragma unroll
    for (int j = 0; j < 4; ++j) {
        int c = c0 + j;
        outv[j] = (gv[j] - mu) * inv * g2[c] + n2[c];
        h[(size_t)nn * HID + c] = outv[j];
    }
    unsigned int* hb32 = (unsigned int*)hb;
    size_t p = ((size_t)nn * HID + c0) >> 1;
    hb32[p]     = pack_bf16x2(outv[0], outv[1]);
    hb32[p + 1] = pack_bf16x2(outv[2], outv[3]);
}

// ---------- pooling gate scalar + per-graph max ----------
__global__ __launch_bounds__(256) void gate_dot(const float* __restrict__ g1buf,
                                                const float* __restrict__ Wg2,
                                                const float* __restrict__ bg2,
                                                const int* __restrict__ batch,
                                                float* __restrict__ gate,
                                                float* __restrict__ gm, int n) {
    int nn = blockIdx.x * 8 + (threadIdx.x >> 5);
    int lane = threadIdx.x & 31;
    if (nn >= n) return;
    float d = 0.f;
#pragma unroll
    for (int j = 0; j < 4; ++j) {
        int c = lane * 4 + j;
        d += g1buf[(size_t)nn * HID + c] * Wg2[c];
    }
    d = wave_sum(d);
    if (lane == 0) {
        d += bg2[0];
        gate[nn] = d;
        atomic_max_f32(&gm[batch[nn]], d);
    }
}
__global__ __launch_bounds__(256) void pool_exp(float* __restrict__ gate,
                                                const float* __restrict__ gm,
                                                float* __restrict__ gs,
                                                const int* __restrict__ batch, int n) {
    int i = blockIdx.x * 256 + threadIdx.x;
    if (i >= n) return;
    int b = batch[i];
    float e = __expf(gate[i] - gm[b]);
    gate[i] = e;
    atomicAdd(&gs[b], e);
}
__global__ __launch_bounds__(256) void pool_acc(const float* __restrict__ gate,
                                                const float* __restrict__ gs,
                                                const int* __restrict__ batch,
                                                const float* __restrict__ h,
                                                float* __restrict__ pooled, int n) {
    int nn = blockIdx.x * 8 + (threadIdx.x >> 5);
    int lane = threadIdx.x & 31;
    if (nn >= n) return;
    int b = batch[nn];
    float w = gate[nn] / gs[b];
    int c0 = lane * 4;
#pragma unroll
    for (int j = 0; j < 4; ++j)
        atomicAdd(&pooled[b * HID + c0 + j], w * h[(size_t)nn * HID + c0 + j]);
}

// ---------- final head: out[g,0:2] = k1[g]·Wo2 + bo2 ----------
__global__ __launch_bounds__(256) void final_out(const float* __restrict__ k1,
                                                 const float* __restrict__ Wo2,
                                                 const float* __restrict__ bo2,
                                                 float* __restrict__ out) {
    int g = blockIdx.x * 8 + (threadIdx.x >> 5);
    int lane = threadIdx.x & 31;
    if (g >= NG) return;
    float d0 = 0.f, d1 = 0.f;
#pragma unroll
    for (int j = 0; j < 4; ++j) {
        int c = lane * 4 + j;
        float hv = k1[g * HID + c];
        d0 += hv * Wo2[c * 2 + 0];
        d1 += hv * Wo2[c * 2 + 1];
    }
    d0 = wave_sum(d0); d1 = wave_sum(d1);
    if (lane == 0) { out[g * 2 + 0] = d0 + bo2[0]; out[g * 2 + 1] = d1 + bo2[1]; }
}

// ===================== host launcher =====================
extern "C" void kernel_launch(void* const* d_in, const int* in_sizes, int n_in,
                              void* d_out, int out_size, void* d_ws, size_t ws_size,
                              hipStream_t stream) {
    (void)in_sizes; (void)n_in; (void)out_size; (void)ws_size;

    const float* x     = (const float*)d_in[0];
    const int*   src   = (const int*)d_in[1];
    const int*   dstp  = src + NE;
    const int*   batch = (const int*)d_in[2];
    const float* Win   = (const float*)d_in[3];
    const float* binp  = (const float*)d_in[4];
    auto LP = [&](int l, int j) { return (const float*)d_in[5 + l * 17 + j]; };
    const float* Wg1 = (const float*)d_in[107];
    const float* bg1 = (const float*)d_in[108];
    const float* Wg2 = (const float*)d_in[109];
    const float* bg2 = (const float*)d_in[110];
    const float* Wo1 = (const float*)d_in[111];
    const float* bo1 = (const float*)d_in[112];
    const float* Wo2 = (const float*)d_in[113];
    const float* bo2 = (const float*)d_in[114];

    char* wp = (char*)d_ws;
    auto alloc = [&](size_t bytes) -> void* {
        void* p = wp; wp += (bytes + 255) & ~(size_t)255; return p;
    };
    float*          h      = (float*)alloc((size_t)NN * HID * 4);
    unsigned short* hb     = (unsigned short*)alloc((size_t)NN * HID * 2);   // bf16 mirror
    unsigned short* qb     = (unsigned short*)alloc((size_t)NN * 1024 * 2);
    unsigned short* kb     = (unsigned short*)alloc((size_t)NN * 1024 * 2);
    unsigned short* vb     = (unsigned short*)alloc((size_t)NN * 1024 * 2);
    float*          xr     = (float*)alloc((size_t)NN * HID * 4);
    float*          alpha  = (float*)alloc((size_t)NE * 8 * 4);
    float*          mbuf   = (float*)alloc((size_t)NN * 8 * 4);
    float*          sbuf   = (float*)alloc((size_t)NN * 8 * 4);
    float*          attn   = (float*)alloc((size_t)NN * HID * 4);
    unsigned short* midb   = (unsigned short*)alloc((size_t)NN * 256 * 2);   // FFN mid bf16
    float*          fbuf   = (float*)alloc((size_t)NN * HID * 4);            // FFN out / gate1
    unsigned short* wbuf   = (unsigned short*)alloc((size_t)128 * 1024 * 2); // B^T bf16
    float*          gate   = (float*)alloc((size_t)NN * 4);
    float*          gm     = (float*)alloc(NG * 4);
    float*          gs     = (float*)alloc(NG * 4);
    float*          pooled = (float*)alloc(NG * HID * 4);
    unsigned short* pooledb= (unsigned short*)alloc(NG * HID * 2);
    float*          k1     = (float*)alloc(NG * HID * 4);

    const float NEG_INF = -INFINITY;
    auto fill = [&](float* p, float v, int n) {
        fill_f32<<<(n + 255) / 256, 256, 0, stream>>>(p, v, n);
    };
    auto castT = [&](const float* s, int kshift, int Ncols) {  // -> wbuf = W^T bf16 [N,K]
        int n = Ncols << kshift;
        cast_tr_bf16<<<(n + 255) / 256, 256, 0, stream>>>(s, wbuf, kshift, Ncols);
    };
    auto gemm = [&](const unsigned short* A, const float* bias, void* C,
                    int M, int K, int Ncols, int outbf, int act) {
        dim3 g(Ncols / 128, (M + 31) / 32), b(256);
        if (K == 128) {
            if (outbf)         gemm_wmma<128, 1, 0><<<g, b, 0, stream>>>(A, wbuf, bias, C, M, Ncols);
            else if (act == 0) gemm_wmma<128, 0, 0><<<g, b, 0, stream>>>(A, wbuf, bias, C, M, Ncols);
            else if (act == 1) gemm_wmma<128, 0, 1><<<g, b, 0, stream>>>(A, wbuf, bias, C, M, Ncols);
            else               gemm_wmma<128, 0, 2><<<g, b, 0, stream>>>(A, wbuf, bias, C, M, Ncols);
        } else {               // K == 256 (FFN second GEMM, f32 out, no act)
            gemm_wmma<256, 0, 0><<<g, b, 0, stream>>>(A, wbuf, bias, C, M, Ncols);
        }
    };

    // input projection (writes f32 h + bf16 hb)
    input_proj<<<(NN * HID + 255) / 256, 256, 0, stream>>>(x, Win, binp, h, hb, NN);

    for (int l = 0; l < LAYERS; ++l) {
        // q,k,v projections (bf16 A, bf16 out)
        castT(LP(l, 0), 7, 1024);  gemm(hb, LP(l, 1), qb, NN, 128, 1024, 1, 0);
        castT(LP(l, 2), 7, 1024);  gemm(hb, LP(l, 3), kb, NN, 128, 1024, 1, 0);
        castT(LP(l, 4), 7, 1024);  gemm(hb, LP(l, 5), vb, NN, 128, 1024, 1, 0);
        // skip projection (f32 out)
        castT(LP(l, 6), 7, 128);   gemm(hb, LP(l, 7), xr, NN, 128, 128, 0, 0);
        // edge attention (segment softmax over dst)
        fill(mbuf, NEG_INF, NN * 8);
        fill(sbuf, 0.0f, NN * 8);
        fill(attn, 0.0f, NN * HID);
        edge_logits<<<(NE + 7) / 8, 256, 0, stream>>>(qb, kb, src, dstp, alpha, mbuf, NE);
        edge_exp<<<(NE * 8 + 255) / 256, 256, 0, stream>>>(alpha, mbuf, sbuf, dstp, NE);
        edge_scatter<<<(NE + 7) / 8, 256, 0, stream>>>(alpha, sbuf, vb, src, dstp, attn, NE);
        // beta gate + gelu + residual + LN (updates h and hb in-place)
        beta_ln<<<(NN + 7) / 8, 256, 0, stream>>>(attn, xr, LP(l, 8), LP(l, 9), LP(l, 10), h, hb, NN);
        // FFN: mid = h@W1+b in bf16 (bf16 path runs ACT=0, no gelu), f = mid@W2+b in f32
        castT(LP(l, 11), 7, 256);  gemm(hb, LP(l, 12), midb, NN, 128, 256, 1, 1);
        castT(LP(l, 13), 8, 128);  gemm(midb, LP(l, 14), fbuf, NN, 256, 128, 0, 0);
        residual_ln<<<(NN + 7) / 8, 256, 0, stream>>>(fbuf, LP(l, 15), LP(l, 16), h, hb, NN);
    }

    // attentional pooling
    fill(gm, NEG_INF, NG);
    fill(gs, 0.0f, NG);
    fill(pooled, 0.0f, NG * HID);
    castT(Wg1, 7, 128);  gemm(hb, bg1, fbuf, NN, 128, 128, 0, 2);                    // relu
    gate_dot<<<(NN + 7) / 8, 256, 0, stream>>>(fbuf, Wg2, bg2, batch, gate, gm, NN);
    pool_exp<<<(NN + 255) / 256, 256, 0, stream>>>(gate, gm, gs, batch, NN);
    pool_acc<<<(NN + 7) / 8, 256, 0, stream>>>(gate, gs, batch, h, pooled, NN);
    // output head
    cast_bf16<<<(NG * HID + 255) / 256, 256, 0, stream>>>(pooled, pooledb, NG * HID);
    castT(Wo1, 7, 128);  gemm(pooledb, bo1, k1, NG, 128, 128, 0, 1);                 // gelu
    final_out<<<8, 256, 0, stream>>>(k1, Wo2, bo2, (float*)d_out);
}
